// FlashAttention_86002425135546
// MI455X (gfx1250) — matrix operand order repaired and checked by reading, not compiled, not run on hardware
//
#include <hip/hip_runtime.h>

// ---------------------------------------------------------------------------
// CDNA5 (gfx1250) attention block: QKV proj -> flash attention -> out proj.
// Matmuls via v_wmma_f32_16x16x32_bf16 (fp32 -> bf16 on the fly, f32 accum).
// bf16 tile staging uses GLOBAL_LOAD_ASYNC_TO_LDS_B128 (ASYNCcnt), batched
// with immediate IOFFSETs (offset applies to both LDS and global address).
// K is written d-major by its projection GEMM so attention needs no runtime
// transpose. Wave32, 8 waves / 256-thread block.
// ---------------------------------------------------------------------------

typedef __attribute__((ext_vector_type(16))) __bf16 v16bf;
typedef __attribute__((ext_vector_type(8)))  float  v8f;

union FragU {
    uint4 q[2];
    unsigned int u[8];
    v16bf v;
    unsigned short h[16];
};

__device__ __forceinline__ unsigned short f2bf(float f) {
    unsigned int u = __float_as_uint(f);
    unsigned int r = u + 0x7FFFu + ((u >> 16) & 1u);   // round-to-nearest-even
    return (unsigned short)(r >> 16);
}

// Packed f32x2 -> bf16x2 (v_cvt_pk_bf16_f32 when available).
#if __has_builtin(__builtin_amdgcn_cvt_pk_bf16_f32)
typedef __attribute__((ext_vector_type(2))) __bf16 v2bf;
__device__ __forceinline__ unsigned int f2bf_pk(float lo, float hi) {
    union { v2bf v; unsigned int u; } cv;
    cv.v = __builtin_amdgcn_cvt_pk_bf16_f32(lo, hi);
    return cv.u;
}
#else
__device__ __forceinline__ unsigned int f2bf_pk(float lo, float hi) {
    return (unsigned int)f2bf(lo) | ((unsigned int)f2bf(hi) << 16);
}
#endif

__device__ __forceinline__ v8f vzero8() {
    v8f z;
#pragma unroll
    for (int e = 0; e < 8; ++e) z[e] = 0.0f;
    return z;
}

__device__ __forceinline__ v8f wmma_bf16(v16bf a, v16bf b, v8f c) {
    return __builtin_amdgcn_wmma_f32_16x16x32_bf16(false, a, false, b,
                                                   (short)0, c, false, false);
}

// Async copies global -> LDS (GLOBAL_LOAD_ASYNC_TO_LDS_B128, ASYNCcnt).
// LDS byte address = low 32 bits of the flat pointer (ISA 10.2 aperture rule).
// IOFFSET is added to BOTH the LDS and the global address, so consecutive
// 16B chunks need only one base address pair.
__device__ __forceinline__ void async_copy2_b128(const unsigned short* gsrc,
                                                 unsigned short* ldst) {
    unsigned int lds = (unsigned int)(unsigned long long)ldst;
    asm volatile("global_load_async_to_lds_b128 %0, %1, off\n\t"
                 "global_load_async_to_lds_b128 %0, %1, off offset:16"
                 :: "v"(lds), "v"(gsrc)
                 : "memory");
}

__device__ __forceinline__ void async_copy4_b128(const unsigned short* gsrc,
                                                 unsigned short* ldst) {
    unsigned int lds = (unsigned int)(unsigned long long)ldst;
    asm volatile("global_load_async_to_lds_b128 %0, %1, off\n\t"
                 "global_load_async_to_lds_b128 %0, %1, off offset:16\n\t"
                 "global_load_async_to_lds_b128 %0, %1, off offset:32\n\t"
                 "global_load_async_to_lds_b128 %0, %1, off offset:48"
                 :: "v"(lds), "v"(gsrc)
                 : "memory");
}

__device__ __forceinline__ void wait_async0() {
    asm volatile("s_wait_asynccnt 0x0" ::: "memory");
}

// A-fragment (16x32 bf16): lane -> M row (lane&15); halves -> K
__device__ __forceinline__ v16bf ld_frag_a(const unsigned short* base, int stride,
                                           int row, int k0, int lane) {
    int r  = row + (lane & 15);
    int kb = k0 + ((lane >> 4) << 3);
    FragU f;
    f.q[0] = *(const uint4*)(base + r * stride + kb);
    f.q[1] = *(const uint4*)(base + r * stride + kb + 16);
    return f.v;
}

// B-fragment (32x16 bf16): lane -> N col (lane&15); halves -> K (same order as A).
__device__ __forceinline__ v16bf ld_frag_b(const unsigned short* base, int stride,
                                           int k0, int n0, int lane) {
    int kb = k0 + ((lane >> 4) << 3);
    const unsigned short* p = base + kb * stride + n0 + (lane & 15);
    FragU f;
#pragma unroll
    for (int s = 0; s < 8; ++s) {
        f.h[s]     = p[s * stride];
        f.h[s + 8] = p[(s + 16) * stride];
    }
    return f.v;
}

// Convert 16 consecutive fp32 to 16 bf16 (packed) into a FragU.
__device__ __forceinline__ FragU cvt16(const float* src) {
    FragU t;
#pragma unroll
    for (int i = 0; i < 16; i += 4) {
        float4 v4 = *(const float4*)(src + i);
        t.u[i / 2]     = f2bf_pk(v4.x, v4.y);
        t.u[i / 2 + 1] = f2bf_pk(v4.z, v4.w);
    }
    return t;
}

// ---------------------------------------------------------------------------
// GEMM: C[M,N] = A[M,K] @ W[K,N] + bias.  BM=BN=128, BK=32.
//   ABF16   : A is bf16 workspace (async-to-LDS copy) instead of fp32 input
//   OUTMODE : 0 = bf16 row-major  1 = bf16 K-transposed [b,h,d,S]  2 = fp32
// ---------------------------------------------------------------------------
template <int ABF16, int OUTMODE>
__global__ __launch_bounds__(256) void gemm_kernel(const void* __restrict__ Ap,
                                                   const float* __restrict__ W,
                                                   const float* __restrict__ bias,
                                                   void* __restrict__ Cp,
                                                   int M, int N, int K, int S) {
    __shared__ __attribute__((aligned(16))) unsigned short As[128][48];   // 32 k + pad
    __shared__ __attribute__((aligned(16))) unsigned short Ws[32][136];   // 128 n + pad

    const int tid  = threadIdx.x;
    const int lane = tid & 31;
    const int w    = tid >> 5;
    const int m0 = blockIdx.y * 128;
    const int n0 = blockIdx.x * 128;
    const int tmBase = (w & 3) * 32;   // 2 row tiles per wave
    const int tnBase = (w >> 2) * 64;  // 4 col tiles per wave

    v8f acc[2][4];
#pragma unroll
    for (int i = 0; i < 2; ++i)
#pragma unroll
        for (int j = 0; j < 4; ++j) acc[i][j] = vzero8();

    const int ar = tid >> 1, ac0 = (tid & 1) * 16;   // A tile: 128 rows x 32 k
    const int wr = tid >> 3, wc0 = (tid & 7) * 16;   // W tile: 32 rows x 128 n

    for (int kk = 0; kk < K; kk += 32) {
        if (ABF16) {
            const unsigned short* A = (const unsigned short*)Ap;
            async_copy2_b128(A + (size_t)(m0 + ar) * K + kk + ac0, &As[ar][ac0]);
        } else {
            const float* A = (const float*)Ap;
            FragU t = cvt16(A + (size_t)(m0 + ar) * K + kk + ac0);
            *(uint4*)&As[ar][ac0]     = t.q[0];
            *(uint4*)&As[ar][ac0 + 8] = t.q[1];
        }
        {
            FragU t = cvt16(W + (size_t)(kk + wr) * N + n0 + wc0);
            *(uint4*)&Ws[wr][wc0]     = t.q[0];
            *(uint4*)&Ws[wr][wc0 + 8] = t.q[1];
        }
        if (ABF16) wait_async0();
        __syncthreads();

        v16bf a0 = ld_frag_a(&As[0][0], 48, tmBase, 0, lane);
        v16bf a1 = ld_frag_a(&As[0][0], 48, tmBase + 16, 0, lane);
#pragma unroll
        for (int j = 0; j < 4; ++j) {
            v16bf b = ld_frag_b(&Ws[0][0], 136, 0, tnBase + j * 16, lane);
            acc[0][j] = wmma_bf16(a0, b, acc[0][j]);
            acc[1][j] = wmma_bf16(a1, b, acc[1][j]);
        }
        __syncthreads();
    }

    // C/D layout: lane -> N col (lane&15); vgpr v -> M row v + 8*(lane>=16)
    const int cn   = lane & 15;
    const int rsel = (lane >> 4) << 3;
#pragma unroll
    for (int i = 0; i < 2; ++i)
#pragma unroll
        for (int j = 0; j < 4; ++j) {
            int n = n0 + tnBase + j * 16 + cn;
            float bb = bias[n];
#pragma unroll
            for (int v = 0; v < 8; ++v) {
                int m = m0 + tmBase + i * 16 + v + rsel;
                float val = acc[i][j][v] + bb;
                if (OUTMODE == 2) {
                    ((float*)Cp)[(size_t)m * N + n] = val;
                } else if (OUTMODE == 1) {
                    // K stored d-major: [ (b*16+h)*128 + d ][ S ].  B == 2, so
                    // batch split is a compare instead of a division.
                    int b_ = (m >= S) ? 1 : 0;
                    int s_ = m - (b_ ? S : 0);
                    int h_ = n >> 7, d_ = n & 127;
                    ((unsigned short*)Cp)[((size_t)((b_ * 16 + h_) * 128 + d_)) * S + s_] =
                        f2bf(val);
                } else {
                    ((unsigned short*)Cp)[(size_t)m * N + n] = f2bf(val);
                }
            }
        }
}

// ---------------------------------------------------------------------------
// Flash attention: one block per (b, h, 64-row q tile). D = 128, bf16 in,
// f32 accumulate, online softmax over 64-key tiles. K arrives d-major.
// ---------------------------------------------------------------------------
__global__ __launch_bounds__(256) void attn_kernel(const unsigned short* __restrict__ Q,
                                                   const unsigned short* __restrict__ Kt,
                                                   const unsigned short* __restrict__ V,
                                                   unsigned short* __restrict__ Ob,
                                                   int S) {
    __shared__ __attribute__((aligned(16))) unsigned short Qs[64][136];   // q x d
    __shared__ __attribute__((aligned(16))) unsigned short Kts[128][72];  // d x key
    __shared__ __attribute__((aligned(16))) unsigned short Vs[64][136];   // key x d
    __shared__ __attribute__((aligned(16))) unsigned short Ps[64][72];    // q x key (probs)
    __shared__ __attribute__((aligned(16))) float Ss[64][68];             // q x key scores
    __shared__ float row_m[64], row_l[64], row_c[64];

    const int tid  = threadIdx.x;
    const int lane = tid & 31;
    const int w    = tid >> 5;

    const int nqt = S >> 6;
    const int qt  = blockIdx.x % nqt;
    const int bh  = blockIdx.x / nqt;
    const int h   = bh & 15;
    const int b   = bh >> 4;

    const int rowg0 = b * S + qt * 64;
    const int colh  = h * 128;

    // async-load Q tile (bf16 copy): 64 consecutive bytes per thread
    {
        int r = tid >> 2, d0 = (tid & 3) * 32;
        async_copy4_b128(Q + (size_t)(rowg0 + r) * 2048 + colh + d0, &Qs[r][d0]);
    }
    if (tid < 64) { row_m[tid] = -3.0e38f; row_l[tid] = 0.0f; }

    v8f oacc[4];
#pragma unroll
    for (int j = 0; j < 4; ++j) oacc[j] = vzero8();

    const int stm = (w & 3) * 16, stn0 = (w >> 2) * 32;   // score tiles per wave
    const int om  = (w & 3) * 16, on0  = (w >> 2) * 64;   // O tiles per wave
    const int cn   = lane & 15;
    const int rsel = (lane >> 4) << 3;
    const float scale = 0.08838834764831845f;             // 1/sqrt(128)

    wait_async0();
    __syncthreads();

    const unsigned short* KtH = Kt + (size_t)((b * 16 + h) * 128) * S;

    for (int kt = 0; kt < nqt; ++kt) {
        // async-load K tile (already d-major: rows d, cols key) and V tile
        {
            int dRow = tid >> 1, kc0 = (tid & 1) * 32;
            async_copy4_b128(KtH + (size_t)dRow * S + kt * 64 + kc0, &Kts[dRow][kc0]);

            int r = tid >> 2, d0 = (tid & 3) * 32;
            async_copy4_b128(V + (size_t)(b * S + kt * 64 + r) * 2048 + colh + d0,
                             &Vs[r][d0]);
        }
        wait_async0();
        __syncthreads();

        // scores: S = Q @ K^T  (wave owns 1x2 of 16x16 tiles)
        v8f sacc[2];
        sacc[0] = vzero8(); sacc[1] = vzero8();
#pragma unroll
        for (int dk = 0; dk < 4; ++dk) {
            v16bf aq = ld_frag_a(&Qs[0][0], 136, stm, dk * 32, lane);
#pragma unroll
            for (int j = 0; j < 2; ++j) {
                v16bf bk = ld_frag_b(&Kts[0][0], 72, dk * 32, stn0 + j * 16, lane);
                sacc[j] = wmma_bf16(aq, bk, sacc[j]);
            }
        }
#pragma unroll
        for (int j = 0; j < 2; ++j)
#pragma unroll
            for (int v = 0; v < 8; ++v)
                Ss[stm + v + rsel][stn0 + j * 16 + cn] = sacc[j][v] * scale;
        __syncthreads();

        // online softmax (threads 0..63, one row each)
        if (tid < 64) {
            float mo = row_m[tid];
            float mx = mo;
            for (int jj = 0; jj < 64; ++jj) mx = fmaxf(mx, Ss[tid][jj]);
            float c  = __expf(mo - mx);
            float ls = row_l[tid] * c;
            for (int jj = 0; jj < 64; jj += 2) {
                float p0 = __expf(Ss[tid][jj]     - mx);
                float p1 = __expf(Ss[tid][jj + 1] - mx);
                ls += p0 + p1;
                *(unsigned int*)&Ps[tid][jj] = f2bf_pk(p0, p1);
            }
            row_m[tid] = mx; row_l[tid] = ls; row_c[tid] = c;
        }
        __syncthreads();

        // rescale O then accumulate P @ V  (wave owns 1x4 of 16x16 O tiles)
#pragma unroll
        for (int j = 0; j < 4; ++j)
#pragma unroll
            for (int v = 0; v < 8; ++v)
                oacc[j][v] *= row_c[om + v + rsel];
#pragma unroll
        for (int dk = 0; dk < 2; ++dk) {
            v16bf ap = ld_frag_a(&Ps[0][0], 72, om, dk * 32, lane);
#pragma unroll
            for (int j = 0; j < 4; ++j) {
                v16bf bv = ld_frag_b(&Vs[0][0], 136, dk * 32, on0 + j * 16, lane);
                oacc[j] = wmma_bf16(ap, bv, oacc[j]);
            }
        }
        __syncthreads();   // before next iteration overwrites K/V/S/P tiles
    }

    // normalize and store (token-major bf16 for final projection GEMM)
#pragma unroll
    for (int j = 0; j < 4; ++j)
#pragma unroll
        for (int v = 0; v < 8; ++v) {
            int rl = om + v + rsel;
            float inv = 1.0f / row_l[rl];
            int d = on0 + j * 16 + cn;
            Ob[(size_t)(rowg0 + rl) * 2048 + colh + d] = f2bf(oacc[j][v] * inv);
        }
}

// ---------------------------------------------------------------------------
extern "C" void kernel_launch(void* const* d_in, const int* in_sizes, int n_in,
                              void* d_out, int out_size, void* d_ws, size_t ws_size,
                              hipStream_t stream) {
    const float* hs = (const float*)d_in[0];
    const float* wq = (const float*)d_in[1];
    const float* bq = (const float*)d_in[2];
    const float* wk = (const float*)d_in[3];
    const float* bk = (const float*)d_in[4];
    const float* wv = (const float*)d_in[5];
    const float* bv = (const float*)d_in[6];
    const float* wo = (const float*)d_in[7];
    const float* bo = (const float*)d_in[8];

    const int HID = 2048;
    const int M = in_sizes[0] / HID;   // B*S = 4096
    const int B = 2;
    const int S = M / B;               // 2048

    size_t nelem = (size_t)M * HID;
    unsigned short* Qb = (unsigned short*)d_ws;
    unsigned short* Kb = Qb + nelem;   // d-major: [b,h,d,S]
    unsigned short* Vb = Kb + nelem;
    unsigned short* Ab = Vb + nelem;   // attention output (bf16, token-major)

    dim3 blk(256);
    dim3 gg(HID / 128, M / 128);

    // QKV projections (fp32 in -> bf16 out); K written d-major
    gemm_kernel<0, 0><<<gg, blk, 0, stream>>>(hs, wq, bq, Qb, M, HID, HID, S);
    gemm_kernel<0, 1><<<gg, blk, 0, stream>>>(hs, wk, bk, Kb, M, HID, HID, S);
    gemm_kernel<0, 0><<<gg, blk, 0, stream>>>(hs, wv, bv, Vb, M, HID, HID, S);

    // flash attention: one block per (b, h, 64-row q tile)
    dim3 ga(B * 16 * (S / 64));
    attn_kernel<<<ga, blk, 0, stream>>>(Qb, Kb, Vb, Ab, S);

    // output projection (bf16 in -> fp32 out)
    gemm_kernel<1, 2><<<gg, blk, 0, stream>>>(Ab, wo, bo, (float*)d_out, M, HID, HID, S);
}
